// MPNN_10694468567644
// MI455X (gfx1250) — matrix operand order repaired and checked by reading, not compiled, not run on hardware
//
#include <hip/hip_runtime.h>
#include <hip/hip_bf16.h>

typedef __attribute__((ext_vector_type(16))) _Float16 v16h;
typedef __attribute__((ext_vector_type(8)))  _Float16 v8h;
typedef __attribute__((ext_vector_type(8)))  float    v8f;

#define H_     256
#define NB_    64
#define NA_    30
#define AF_    39      // atom feature dim
#define BF_    50      // bond feature dim (39+11)
#define MSG_S  264     // message row stride (halfs), mult of 8, padded vs 256
#define AI_S   328     // ainput row stride (halfs), K padded 296->320, +8 pad
#define FB_S   72      // staged fbond row stride (halfs), K padded 50->64

#if __has_builtin(__builtin_amdgcn_sched_barrier)
#define SCHED_FENCE() __builtin_amdgcn_sched_barrier(0)
#else
#define SCHED_FENCE()
#endif

// ---- LDS layout (bytes) ----  (binput now lives in registers)
#define OFF_MSG   0
#define SZ_MSG    (NB_*MSG_S*2)            // 33792
#define OFF_NEI   (OFF_MSG+SZ_MSG)
#define SZ_NEI    (NB_*MSG_S*2)            // 33792 (aliases: fbond stage, ainput)
#define OFF_OACC  (OFF_NEI+SZ_NEI)
#define SZ_OACC   (H_*4)
#define OFF_BG    (OFF_OACC+SZ_OACC)
#define SZ_BG     (NB_*6*4)
#define OFF_BO    (OFF_BG+SZ_BG)
#define SZ_BO     (H_*4)
#define LDS_BYTES (OFF_BO+SZ_BO)           // 71168 -> up to 4 workgroups / WGP

// ---- packed weight sizes (halfs). B-tile = 32x16 = 512 halfs ----
#define WI_HALFS  (2*16*512)    // K 64,  N 256
#define WH_HALFS  (8*16*512)    // K 256, N 256
#define WO_HALFS  (10*16*512)   // K 320, N 256
#define PACK_TOTAL (WI_HALFS+WH_HALFS+WO_HALFS)   // 163840 halfs = 320KB ws

__device__ __forceinline__ v8f zero_v8f() {
  v8f r;
#pragma unroll
  for (int i = 0; i < 8; ++i) r[i] = 0.f;
  return r;
}
__device__ __forceinline__ v8h zero_v8h() {
  v8h r;
#pragma unroll
  for (int i = 0; i < 8; ++i) r[i] = (_Float16)0.f;
  return r;
}

// A-fragment (16x32 f16): lane holds row=lane&15, halfs {k0..k0+7} and {k0+16..k0+23}
__device__ __forceinline__ v16h load_a16(const _Float16* p) {
  v8h lo = *(const v8h*)p;
  v8h hi = *(const v8h*)(p + 16);
  return __builtin_shufflevector(lo, hi, 0,1,2,3,4,5,6,7,8,9,10,11,12,13,14,15);
}

__device__ __forceinline__ v8f wmma_f16(v16h a, v16h b, v8f c) {
  return __builtin_amdgcn_wmma_f32_16x16x32_f16(false, a, false, b, (short)0, c,
                                                false, false);
}

// ------------------------------------------------------------------
// Repack W_i / W_h / W_o into f16 WMMA B-tile layout:
// tile (kt,nt): lane l holds column n = nt*16 + (l&15), halfs s<8 -> k = 8*(l>>4)+s,
// s>=8 -> k = 16 + 8*(l>>4) + (s-8). Half address:
//   ((kt*16+nt)*32 + lane)*16 + s
// ------------------------------------------------------------------
__global__ void pack_weights_k(const float* __restrict__ Wi,
                               const float* __restrict__ Wh,
                               const float* __restrict__ Wo,
                               _Float16* __restrict__ ws) {
  int gid = blockIdx.x * 256 + threadIdx.x;
  if (gid >= PACK_TOTAL) return;
  int mat, off;
  if (gid < WI_HALFS)                { mat = 0; off = gid; }
  else if (gid < WI_HALFS + WH_HALFS){ mat = 1; off = gid - WI_HALFS; }
  else                               { mat = 2; off = gid - WI_HALFS - WH_HALFS; }
  int tile   = off >> 9;          // 512 halfs / tile
  int within = off & 511;
  int lane   = within >> 4;
  int klocal = within & 15;
  int kt = tile >> 4;
  int nt = tile & 15;
  int k  = kt * 32 + ((klocal >> 3) << 4) + ((lane >> 4) << 3) + (klocal & 7);
  int n  = (nt << 4) + (lane & 15);
  float v = 0.f;
  if (mat == 0) {                        // W_i: [50,256], K padded to 64
    if (k < BF_) v = Wi[k * H_ + n];
  } else if (mat == 1) {                 // W_h: [256,256]
    v = Wh[k * H_ + n];
  } else {                               // W_o: [295,256] with spacer row at k=39
    if (k < AF_)                        v = Wo[k * H_ + n];
    else if (k >= 40 && k <= AF_ + H_)  v = Wo[(k - 1) * H_ + n];
  }
  ws[gid] = (_Float16)v;
}

// GEMM1 chunk: 2 K-steps, 4 N-tiles; produces binput (f16 regs) + msg=relu
#define GEMM1_CHUNK(CIDX, BINH)                                               \
  do {                                                                        \
    const int nb = ntb + (CIDX) * 4;                                          \
    v16h b[2][4];                                                             \
    _Pragma("unroll") for (int kt = 0; kt < 2; ++kt)                          \
      _Pragma("unroll") for (int j = 0; j < 4; ++j)                           \
        b[kt][j] = bp[(kt * 16 + nb + j) * 32];                               \
    v8f acc[4];                                                               \
    _Pragma("unroll") for (int j = 0; j < 4; ++j) acc[j] = zero_v8f();        \
    _Pragma("unroll") for (int kt = 0; kt < 2; ++kt)                          \
      _Pragma("unroll") for (int j = 0; j < 4; ++j)                           \
        acc[j] = wmma_f16(a[kt], b[kt][j], acc[j]);                           \
    _Pragma("unroll") for (int j = 0; j < 4; ++j) {                           \
      const int n = (nb + j) * 16 + lrow;                                     \
      _Pragma("unroll") for (int jj = 0; jj < 8; ++jj) {                      \
        const int m = mtile * 16 + lhi * 8 + jj;                              \
        float v = acc[j][jj];                                                 \
        BINH[j][jj] = (_Float16)v;                                            \
        msg[m * MSG_S + n] = (_Float16)fmaxf(v, 0.f);                         \
      }                                                                       \
    }                                                                         \
  } while (0)

// GEMM2 chunk: 8 K-steps (B double-buffered), 4 N-tiles; msg=relu(binput+acc)
#define GEMM2_CHUNK(CIDX, BINH)                                               \
  do {                                                                        \
    const int nb = ntb + (CIDX) * 4;                                          \
    v8f acc[4];                                                               \
    _Pragma("unroll") for (int j = 0; j < 4; ++j) acc[j] = zero_v8f();        \
    v16h bcur[4], bnxt[4];                                                    \
    _Pragma("unroll") for (int j = 0; j < 4; ++j) bcur[j] = bp[(nb + j) * 32];\
    _Pragma("unroll") for (int kt = 0; kt < 8; ++kt) {                        \
      if (kt < 7) {                                                           \
        _Pragma("unroll") for (int j = 0; j < 4; ++j)                         \
          bnxt[j] = bp[((kt + 1) * 16 + nb + j) * 32];                        \
      }                                                                       \
      _Pragma("unroll") for (int j = 0; j < 4; ++j)                           \
        acc[j] = wmma_f16(a[kt], bcur[j], acc[j]);                            \
      _Pragma("unroll") for (int j = 0; j < 4; ++j) bcur[j] = bnxt[j];        \
    }                                                                         \
    _Pragma("unroll") for (int j = 0; j < 4; ++j) {                           \
      const int n = (nb + j) * 16 + lrow;                                     \
      _Pragma("unroll") for (int jj = 0; jj < 8; ++jj) {                      \
        const int m = mtile * 16 + lhi * 8 + jj;                              \
        float v = (float)BINH[j][jj] + acc[j][jj];                            \
        msg[m * MSG_S + n] = (_Float16)fmaxf(v, 0.f);                         \
      }                                                                       \
    }                                                                         \
  } while (0)

// ------------------------------------------------------------------
// Main kernel: one workgroup per molecule, 256 threads = 8 wave32s.
// ------------------------------------------------------------------
__global__ __launch_bounds__(256) void mpnn_kernel(
    const float* __restrict__ fatoms, const float* __restrict__ fbonds,
    const int* __restrict__ agraph, const int* __restrict__ bgraph,
    const _Float16* __restrict__ pWi, const _Float16* __restrict__ pWh,
    const _Float16* __restrict__ pWo, const float* __restrict__ b_o,
    float* __restrict__ out) {
  extern __shared__ char smem[];
  _Float16* msg  = (_Float16*)(smem + OFF_MSG);
  _Float16* nei  = (_Float16*)(smem + OFF_NEI);   // aliased: fbond stage / ainput
  float*    oacc = (float*)(smem + OFF_OACC);
  int*      bgS  = (int*)(smem + OFF_BG);
  float*    boS  = (float*)(smem + OFF_BO);

  const int t    = threadIdx.x;
  const int mol  = blockIdx.x;
  const int lane = t & 31;
  const int w    = t >> 5;
  const int lrow = lane & 15;
  const int lhi  = lane >> 4;
  const int mtile = w >> 1;          // bond-GEMM M-tile owned by this wave
  const int ntb   = (w & 1) * 8;     // bond-GEMM first N-tile owned by this wave

  // binput tiles, kept in registers for the whole depth loop (f16, 32 VGPRs)
  v8h binh0[4], binh1[4];

  // ---- phase 1: stage fbonds (f16, K padded to 64) into nei buffer; bg, b_o, oacc
  {
    const int row = t >> 2;
    const int c0  = (t & 3) * 16;
    const float* src = fbonds + (size_t)(mol * NB_ + row) * BF_;
#pragma unroll
    for (int c = 0; c < 16; ++c) {
      int cc = c0 + c;
      float v = (cc < BF_) ? src[cc] : 0.f;
      nei[row * FB_S + cc] = (_Float16)v;
    }
    oacc[t] = 0.f;
    boS[t]  = b_o[t];
    for (int i = t; i < NB_ * 6; i += 256) bgS[i] = bgraph[(size_t)mol * NB_ * 6 + i];
  }
  __syncthreads();

  // ---- phase 2: binput = fbonds @ W_i (to registers);  msg = relu(binput)
  {
    const int arow = mtile * 16 + lrow;
    v16h a[2];
#pragma unroll
    for (int kt = 0; kt < 2; ++kt)
      a[kt] = load_a16(&nei[arow * FB_S + kt * 32 + lhi * 8]);
    const v16h* bp = (const v16h*)pWi + lane;
    GEMM1_CHUNK(0, binh0);
    SCHED_FENCE();
    GEMM1_CHUNK(1, binh1);
  }
  __syncthreads();

  // ---- phase 3: 5 message-passing iterations
  for (int depth = 0; depth < 5; ++depth) {
    // gather: nei[b][:] = sum over 6 neighbors of msg
    {
      const int b  = t >> 2;
      const int h0 = (t & 3) * 64;
      const int i0 = bgS[b * 6 + 0] * MSG_S, i1 = bgS[b * 6 + 1] * MSG_S,
                i2 = bgS[b * 6 + 2] * MSG_S, i3 = bgS[b * 6 + 3] * MSG_S,
                i4 = bgS[b * 6 + 4] * MSG_S, i5 = bgS[b * 6 + 5] * MSG_S;
#pragma unroll
      for (int hh = 0; hh < 64; hh += 8) {
        const int h = h0 + hh;
        v8h s = *(const v8h*)&msg[i0 + h];
        s = s + *(const v8h*)&msg[i1 + h];
        s = s + *(const v8h*)&msg[i2 + h];
        s = s + *(const v8h*)&msg[i3 + h];
        s = s + *(const v8h*)&msg[i4 + h];
        s = s + *(const v8h*)&msg[i5 + h];
        *(v8h*)&nei[b * MSG_S + h] = s;
      }
    }
    __syncthreads();
    // msg = relu(binput + nei @ W_h)
    {
      const int arow = mtile * 16 + lrow;
      v16h a[8];
#pragma unroll
      for (int kt = 0; kt < 8; ++kt)
        a[kt] = load_a16(&nei[arow * MSG_S + kt * 32 + lhi * 8]);
      const v16h* bp = (const v16h*)pWh + lane;
      GEMM2_CHUNK(0, binh0);
      SCHED_FENCE();
      GEMM2_CHUNK(1, binh1);
    }
    __syncthreads();
  }

  // ---- phase 4: build ainput = [fa(39) | 0 | bond->atom nei(256) | pad] in nei buf
  {
    v8h z = zero_v8h();
    for (int i = t; i < (32 * AI_S) / 8; i += 256) ((v8h*)nei)[i] = z;
  }
  __syncthreads();
  {
    for (int i = t; i < NA_ * AF_; i += 256) {
      int r = i / AF_, c = i - r * AF_;
      nei[r * AI_S + c] = (_Float16)fatoms[(size_t)(mol * NA_ + r) * AF_ + c];
    }
    const int atom = t >> 3;
    const int h0   = (t & 7) * 32;
    if (atom < NA_) {
      const int* ag = agraph + (size_t)(mol * NA_ + atom) * 6;
      const int i0 = ag[0] * MSG_S, i1 = ag[1] * MSG_S, i2 = ag[2] * MSG_S,
                i3 = ag[3] * MSG_S, i4 = ag[4] * MSG_S, i5 = ag[5] * MSG_S;
#pragma unroll
      for (int hh = 0; hh < 32; hh += 8) {
        const int h = h0 + hh;
        v8h s = *(const v8h*)&msg[i0 + h];
        s = s + *(const v8h*)&msg[i1 + h];
        s = s + *(const v8h*)&msg[i2 + h];
        s = s + *(const v8h*)&msg[i3 + h];
        s = s + *(const v8h*)&msg[i4 + h];
        s = s + *(const v8h*)&msg[i5 + h];
        *(v8h*)&nei[atom * AI_S + 40 + h] = s;   // nei block starts at col 40
      }
    }
  }
  __syncthreads();

  // ---- phase 5: atom_hiddens = relu(ainput @ W_o + b_o); masked column sums
  {
    const int amt  = w >> 2;           // 0..1
    const int antb = (w & 3) * 4;      // 4 N-tiles per wave
    const int arow = amt * 16 + lrow;
    v16h a[10];
#pragma unroll
    for (int kt = 0; kt < 10; ++kt)
      a[kt] = load_a16(&nei[arow * AI_S + kt * 32 + lhi * 8]);
    const v16h* bp = (const v16h*)pWo + lane;
    v8f acc[4];
#pragma unroll
    for (int j = 0; j < 4; ++j) acc[j] = zero_v8f();
    v16h bcur[4], bnxt[4];
#pragma unroll
    for (int j = 0; j < 4; ++j) bcur[j] = bp[(antb + j) * 32];
#pragma unroll
    for (int kt = 0; kt < 10; ++kt) {
      if (kt < 9) {
#pragma unroll
        for (int j = 0; j < 4; ++j) bnxt[j] = bp[((kt + 1) * 16 + antb + j) * 32];
      }
#pragma unroll
      for (int j = 0; j < 4; ++j) acc[j] = wmma_f16(a[kt], bcur[j], acc[j]);
#pragma unroll
      for (int j = 0; j < 4; ++j) bcur[j] = bnxt[j];
    }
#pragma unroll
    for (int j = 0; j < 4; ++j) {
      const int n    = (antb + j) * 16 + lrow;
      const float bb = boS[n];
      float part = 0.f;
#pragma unroll
      for (int jj = 0; jj < 8; ++jj) {
        const int m = amt * 16 + lhi * 8 + jj;
        float v = fmaxf(acc[j][jj] + bb, 0.f);
        part += (m < NA_) ? v : 0.f;     // exclude padded rows 30,31
      }
      atomicAdd(&oacc[n], part);
    }
  }
  __syncthreads();
  out[(size_t)mol * H_ + t] = oacc[t] * (1.0f / 30.0f);
}

extern "C" void kernel_launch(void* const* d_in, const int* in_sizes, int n_in,
                              void* d_out, int out_size, void* d_ws, size_t ws_size,
                              hipStream_t stream) {
  const float* fatoms = (const float*)d_in[0];
  const float* fbonds = (const float*)d_in[1];
  const int*   agraph = (const int*)d_in[2];
  const int*   bgraph = (const int*)d_in[3];
  const float* Wi     = (const float*)d_in[4];
  const float* Wh     = (const float*)d_in[5];
  const float* Wo     = (const float*)d_in[6];
  const float* bo     = (const float*)d_in[7];
  float* out = (float*)d_out;
  _Float16* pack = (_Float16*)d_ws;   // 320KB packed f16 weights

  const int Bmol = in_sizes[0] / (NA_ * AF_);   // 1024

  pack_weights_k<<<(PACK_TOTAL + 255) / 256, 256, 0, stream>>>(Wi, Wh, Wo, pack);

  (void)hipFuncSetAttribute((const void*)mpnn_kernel,
                            hipFuncAttributeMaxDynamicSharedMemorySize, LDS_BYTES);
  mpnn_kernel<<<Bmol, 256, LDS_BYTES, stream>>>(
      fatoms, fbonds, agraph, bgraph,
      pack, pack + WI_HALFS, pack + WI_HALFS + WH_HALFS, bo, out);
}
